// TransformerBlock_6047313953584
// MI455X (gfx1250) — matrix operand order repaired and checked by reading, not compiled, not run on hardware
//
#include <hip/hip_runtime.h>
#include <hip/hip_bf16.h>
#include <math.h>

// ---------------- problem constants ----------------
#define D_MODEL  1024
#define N_HEADS  16
#define HEAD_DIM 64
#define D_FF     4096
#define BATCH    2
#define SEQ      2048
#define MROWS    (BATCH * SEQ)          // 4096
#define EPSV     1e-5f

typedef __attribute__((ext_vector_type(16))) _Float16 v16h;
typedef __attribute__((ext_vector_type(8)))  float    v8f;
typedef __attribute__((ext_vector_type(4)))  unsigned int v4u;
typedef __attribute__((ext_vector_type(8)))  int      v8i;
typedef __attribute__((ext_vector_type(4)))  int      v4i;

#if defined(__has_builtin)
#if __has_builtin(__builtin_amdgcn_tensor_load_to_lds) && __has_builtin(__builtin_amdgcn_s_wait_tensorcnt)
#define USE_TDM 1
#endif
#endif
#ifndef USE_TDM
#define USE_TDM 0
#endif

#if USE_TDM
// 2-D fp16 tile load via Tensor Data Mover: tile_h rows x tile_w cols from a
// row-major [tensor_rows][row_stride] fp16 tensor into contiguous LDS.
__device__ inline void tdm_load_2d(void* lds, const _Float16* gptr,
                                   int row_stride, int tensor_rows,
                                   int tile_w, int tile_h) {
    unsigned ldsa = (unsigned)(size_t)lds;
    unsigned long long ga = (unsigned long long)(size_t)gptr;
    // D# group0: count=1 | lds_addr | global_addr | type=2
    v4u g0 = { 1u, ldsa, (unsigned)ga,
               (unsigned)((ga >> 32) & 0x1FFFFFFu) | (2u << 30) };
    unsigned td0 = (unsigned)row_stride;       // tensor_dim0
    unsigned td1 = (unsigned)tensor_rows;      // tensor_dim1
    // D# group1: data_size=2B; dims, tile dims, dim0 stride
    v8i g1 = { (int)(1u << 16),
               (int)((td0 & 0xFFFFu) << 16),
               (int)(((td0 >> 16) & 0xFFFFu) | ((td1 & 0xFFFFu) << 16)),
               (int)(((td1 >> 16) & 0xFFFFu) | ((unsigned)tile_w << 16)),
               (int)(unsigned)tile_h,
               (int)td0,
               0, 0 };
    v4i gz  = (v4i){};
    v8i gz8 = (v8i){};
    __builtin_amdgcn_tensor_load_to_lds(g0, g1, gz, gz, gz8, 0);
}
#endif

// =====================================================================
// fp32 [R][C]  ->  fp16 transposed [C][R]   (weights, once per launch)
// =====================================================================
__global__ __launch_bounds__(256)
void f32_to_f16_t_kernel(const float* __restrict__ in, _Float16* __restrict__ out,
                         int R, int C) {
    __shared__ float tile[32][33];
    int c0 = blockIdx.x * 32, r0 = blockIdx.y * 32;
    int tx = threadIdx.x & 31, ty = threadIdx.x >> 5;     // ty: 0..7
    #pragma unroll
    for (int i = 0; i < 32; i += 8)
        tile[ty + i][tx] = in[(size_t)(r0 + ty + i) * C + c0 + tx];
    __syncthreads();
    #pragma unroll
    for (int i = 0; i < 32; i += 8)
        out[(size_t)(c0 + ty + i) * R + r0 + tx] = (_Float16)tile[tx][ty + i];
}

// =====================================================================
// time-MLP: h = silu(time_emb @ w1 + b1)            th: [2][B][1024]
// =====================================================================
__global__ void time_hidden_kernel(const float* __restrict__ te,
                                   const float* __restrict__ w1a, const float* __restrict__ b1a,
                                   const float* __restrict__ w1b, const float* __restrict__ b1b,
                                   float* __restrict__ th) {
    int idx = blockIdx.x * blockDim.x + threadIdx.x;      // 2*B*1024
    if (idx >= 2 * BATCH * D_MODEL) return;
    int j   = idx & (D_MODEL - 1);
    int b   = (idx >> 10) & (BATCH - 1);
    int nrm = idx >> 11;
    const float* w1 = nrm ? w1b : w1a;
    const float* b1 = nrm ? b1b : b1a;
    const float* t  = te + (size_t)b * D_MODEL;
    float acc = b1[j];
    for (int i = 0; i < D_MODEL; ++i) acc += t[i] * w1[(size_t)i * D_MODEL + j];
    th[idx] = acc / (1.0f + __expf(-acc));                // silu
}

// ss = th @ w2 + b2 ; ss: [2][B][2048] (scale = [:1024], shift = [1024:])
__global__ void time_ss_kernel(const float* __restrict__ th,
                               const float* __restrict__ w2a, const float* __restrict__ b2a,
                               const float* __restrict__ w2b, const float* __restrict__ b2b,
                               float* __restrict__ ss) {
    int idx = blockIdx.x * blockDim.x + threadIdx.x;      // 2*B*2048
    if (idx >= 2 * BATCH * 2 * D_MODEL) return;
    int j   = idx & (2 * D_MODEL - 1);
    int b   = (idx >> 11) & (BATCH - 1);
    int nrm = idx >> 12;
    const float* w2 = nrm ? w2b : w2a;
    const float* b2 = nrm ? b2b : b2a;
    const float* h  = th + (size_t)(nrm * BATCH + b) * D_MODEL;
    float acc = b2[j];
    for (int i = 0; i < D_MODEL; ++i) acc += h[i] * w2[(size_t)i * 2 * D_MODEL + j];
    ss[idx] = acc;
}

// =====================================================================
// AdaRMSNorm -> fp16 activations.  One 256-thread block per (b,s) row.
// =====================================================================
__global__ __launch_bounds__(256)
void ada_norm_kernel(const float* __restrict__ x, const float* __restrict__ w,
                     const float* __restrict__ ss,   // [B][2048] for this norm
                     _Float16* __restrict__ out) {
    __shared__ float red[8];
    int row = blockIdx.x;                      // b*SEQ + s
    int b   = row / SEQ;
    const float* xr = x + (size_t)row * D_MODEL;

    float sum = 0.0f;
    for (int i = threadIdx.x; i < D_MODEL; i += 256) { float v = xr[i]; sum += v * v; }
    #pragma unroll
    for (int off = 1; off < 32; off <<= 1) sum += __shfl_xor(sum, off, 32);
    if ((threadIdx.x & 31) == 0) red[threadIdx.x >> 5] = sum;
    __syncthreads();
    float tot = 0.0f;
    #pragma unroll
    for (int i = 0; i < 8; ++i) tot += red[i];
    float rn = rsqrtf(tot / (float)D_MODEL + EPSV);

    const float* sc = ss + (size_t)b * 2 * D_MODEL;
    const float* sh = sc + D_MODEL;
    for (int i = threadIdx.x; i < D_MODEL; i += 256) {
        float v = xr[i] * rn * w[i] * (1.0f + sc[i]) + sh[i];
        out[(size_t)row * D_MODEL + i] = (_Float16)v;
    }
}

// =====================================================================
// WMMA GEMM: C[M,N] = A[M,K](f16,row) * W (given as BT[N][K], f16) [+bias]
// 128x128 block tile, BK=32, 8 waves, each wave owns 32x64 (2x4 WMMA tiles).
// Both tiles staged by the Tensor Data Mover, double-buffered: wave 0 issues
// next iteration's two tensor_load_to_lds right after the barrier, waits
// s_wait_tensorcnt==0 at the top of the next iteration (TDM overlaps WMMA).
// MODE 0: outF = acc+bias
// MODE 1: outF = acc+bias + aux[o]              (residual add)
// MODE 2: outH = (f16)(acc+bias)
// MODE 3: outH = (f16)( gelu(aux[o]) * acc )    (fused gate*up)
// =====================================================================
#define BM 128
#define BN 128
#define BK 32

template <int MODE>
__global__ __launch_bounds__(256)
void wmma_gemm_kernel(const _Float16* __restrict__ A, const _Float16* __restrict__ BT,
                      const float* __restrict__ bias, const float* __restrict__ aux,
                      float* __restrict__ outF, _Float16* __restrict__ outH,
                      int M, int N, int K) {
    __shared__ _Float16 As[2][BM * BK];   // [m][k] row-major
    __shared__ _Float16 Bs[2][BN * BK];   // [n][k]  (BT rows are contiguous)

    const int tid  = threadIdx.x;
    const int lane = tid & 31;
    const int wave = tid >> 5;
    const int wm   = wave & 3;         // 32-row group
    const int wn   = wave >> 2;        // 64-col group
    const int half = lane >> 4;
    const int l16  = lane & 15;

    const int m0 = blockIdx.y * BM;
    const int n0 = blockIdx.x * BN;

    v8f acc[2][4];
    #pragma unroll
    for (int i = 0; i < 2; ++i)
        #pragma unroll
        for (int j = 0; j < 4; ++j) acc[i][j] = (v8f){};

    // fragment load + 8 WMMAs from one LDS buffer pair
    auto compute_tiles = [&](const _Float16* Asb, const _Float16* Bsb) {
        v16h bf[4];
        #pragma unroll
        for (int ni = 0; ni < 4; ++ni) {
            const _Float16* p = &Bsb[(wn * 64 + ni * 16 + l16) * BK + 16 * half];
            #pragma unroll
            for (int e = 0; e < 16; ++e) bf[ni][e] = p[e];      // k = e + 16*half
        }
        v16h af[2];
        #pragma unroll
        for (int mi = 0; mi < 2; ++mi) {
            const _Float16* p = &Asb[(wm * 32 + mi * 16 + l16) * BK];
            #pragma unroll
            for (int j = 0; j < 16; ++j) {
                int k = j + 16 * half;                           // A fragment k map
                af[mi][j] = p[k];
            }
        }
        #pragma unroll
        for (int mi = 0; mi < 2; ++mi)
            #pragma unroll
            for (int ni = 0; ni < 4; ++ni)
                acc[mi][ni] = __builtin_amdgcn_wmma_f32_16x16x32_f16(
                    false, af[mi], false, bf[ni], (short)0, acc[mi][ni], false, false);
    };

#if USE_TDM
    const int nsteps = K / BK;
    if (wave == 0) {
        tdm_load_2d(&As[0][0], A  + (size_t)m0 * K, K, M, BK, BM);
        tdm_load_2d(&Bs[0][0], BT + (size_t)n0 * K, K, N, BK, BN);
    }
    for (int i = 0; i < nsteps; ++i) {
        if (wave == 0) __builtin_amdgcn_s_wait_tensorcnt(0);
        __syncthreads();                               // buffer i&1 published
        if (wave == 0 && i + 1 < nsteps) {             // prefetch next via TDM
            int kn = (i + 1) * BK;
            tdm_load_2d(&As[(i + 1) & 1][0], A  + (size_t)m0 * K + kn, K, M, BK, BM);
            tdm_load_2d(&Bs[(i + 1) & 1][0], BT + (size_t)n0 * K + kn, K, N, BK, BN);
        }
        compute_tiles(&As[i & 1][0], &Bs[i & 1][0]);
    }
#else
    for (int k0 = 0; k0 < K; k0 += BK) {
        #pragma unroll
        for (int it = 0; it < 2; ++it) {
            int idx = (tid + it * 256) * 8;             // half-index in 128x32 tile
            int am  = idx >> 5, ak = idx & 31;
            const float4 va = *reinterpret_cast<const float4*>(A + (size_t)(m0 + am) * K + k0 + ak);
            *reinterpret_cast<float4*>(&As[0][am * BK + ak]) = va;
            const float4 vb = *reinterpret_cast<const float4*>(BT + (size_t)(n0 + am) * K + k0 + ak);
            *reinterpret_cast<float4*>(&Bs[0][am * BK + ak]) = vb;
        }
        if (k0 + BK < K) {
            __builtin_prefetch(A  + (size_t)(m0 + (tid >> 1)) * K + k0 + BK + (tid & 1) * 16, 0, 1);
            __builtin_prefetch(BT + (size_t)(n0 + (tid >> 1)) * K + k0 + BK + (tid & 1) * 16, 0, 1);
        }
        __syncthreads();
        compute_tiles(&As[0][0], &Bs[0][0]);
        __syncthreads();
    }
#endif

    // ---- epilogue (C layout: VGPR r -> row r + 8*half, col = lane&15) ----
    #pragma unroll
    for (int mi = 0; mi < 2; ++mi) {
        int rowb = m0 + wm * 32 + mi * 16 + 8 * half;
        #pragma unroll
        for (int ni = 0; ni < 4; ++ni) {
            int col = n0 + wn * 64 + ni * 16 + l16;
            float bv = bias ? bias[col] : 0.0f;
            #pragma unroll
            for (int r = 0; r < 8; ++r) {
                size_t o = (size_t)(rowb + r) * N + col;
                float v = acc[mi][ni][r] + bv;
                if constexpr (MODE == 0) {
                    outF[o] = v;
                } else if constexpr (MODE == 1) {
                    outF[o] = v + aux[o];
                } else if constexpr (MODE == 2) {
                    outH[o] = (_Float16)v;
                } else {
                    float g  = aux[o];
                    float ge = 0.5f * g * (1.0f + erff(g * 0.70710678f));
                    outH[o]  = (_Float16)(ge * v);
                }
            }
        }
    }
}

// =====================================================================
// RoPE: fp32 Q/K -> rotated fp16.  One thread per (b,s,h,freq-pair).
// =====================================================================
__global__ void rope_kernel(const float* __restrict__ q32, const float* __restrict__ k32,
                            _Float16* __restrict__ qh, _Float16* __restrict__ kh, int n) {
    int idx = blockIdx.x * blockDim.x + threadIdx.x;     // B*S*H*32
    if (idx >= n) return;
    int i = idx & 31;
    int t = idx >> 9;                                    // b*SEQ + s
    int s = t % SEQ;
    float f   = __powf(10000.0f, -(float)(2 * i) / 64.0f);
    float ang = (float)s * f;
    float sn, c;
    __sincosf(ang, &sn, &c);
    size_t base = ((size_t)(idx >> 5)) << 6;             // ((b*S+s)*H + h)*64
    float q1 = q32[base + i], q2 = q32[base + 32 + i];
    qh[base + i]      = (_Float16)(q1 * c - q2 * sn);
    qh[base + 32 + i] = (_Float16)(q2 * c + q1 * sn);
    float k1 = k32[base + i], k2 = k32[base + 32 + i];
    kh[base + i]      = (_Float16)(k1 * c - k2 * sn);
    kh[base + 32 + i] = (_Float16)(k2 * c + k1 * sn);
}

// =====================================================================
// Flash attention, fp16 WMMA.  Q/K/V layout [b, s, h, 64].
// grid = (SEQ/64, N_HEADS, BATCH), block = 128 (4 waves x 16 query rows).
// V block staged through LDS transposed so PV B-fragments are contiguous.
// =====================================================================
__global__ __launch_bounds__(128)
void flash_attn_kernel(const _Float16* __restrict__ Q, const _Float16* __restrict__ Km,
                       const _Float16* __restrict__ V, _Float16* __restrict__ O) {
    __shared__ _Float16 Pl[4][16 * 32];                  // per-wave P scratch
    __shared__ _Float16 Vt[HEAD_DIM * 32];               // V block transposed [d][key]

    const int lane = threadIdx.x & 31;
    const int wave = threadIdx.x >> 5;
    const int half = lane >> 4;
    const int l16  = lane & 15;
    const int h    = blockIdx.y;
    const int b    = blockIdx.z;
    const int qm0  = blockIdx.x * 64 + wave * 16;

    // Q fragments: 16 rows x 64 d  ->  two A-frags (d 0..31 / 32..63)
    v16h qf[2];
    {
        const size_t qbase = ((size_t)(b * SEQ + qm0 + l16) * N_HEADS + h) * HEAD_DIM;
        #pragma unroll
        for (int st = 0; st < 2; ++st)
            #pragma unroll
            for (int j = 0; j < 16; ++j) {
                int d = st * 32 + j + 16 * half;
                qf[st][j] = Q[qbase + d];
            }
    }

    v8f o_acc[4];
    #pragma unroll
    for (int ni = 0; ni < 4; ++ni) o_acc[ni] = (v8f){};
    float mrow[8], lrow[8];
    #pragma unroll
    for (int r = 0; r < 8; ++r) { mrow[r] = -1e30f; lrow[r] = 0.0f; }

    for (int t0 = 0; t0 < SEQ; t0 += 32) {
        // ---- cooperative V staging: 32 keys x 64 d -> Vt[d][key] ----
        #pragma unroll
        for (int it = 0; it < 2; ++it) {
            int idx = (threadIdx.x + it * 128) * 8;       // half-index in 32x64 block
            int key = idx >> 6;
            int d8  = idx & 63;
            const size_t vbase = ((size_t)(b * SEQ + t0 + key) * N_HEADS + h) * HEAD_DIM + d8;
            const float4 vv = *reinterpret_cast<const float4*>(V + vbase);
            const _Float16* hp = reinterpret_cast<const _Float16*>(&vv);
            #pragma unroll
            for (int j = 0; j < 8; ++j) Vt[(d8 + j) * 32 + key] = hp[j];
            if (t0 + 32 < SEQ)                            // gfx1250 prefetch next V block
                __builtin_prefetch(V + vbase + (size_t)32 * N_HEADS * HEAD_DIM, 0, 1);
        }
        __syncthreads();

        // ---- scores for two 16-key tiles: S = (Q K^T) * 1/sqrt(64) ----
        v8f s[2];
        #pragma unroll
        for (int kt = 0; kt < 2; ++kt) {
            v8f a = (v8f){};
            #pragma unroll
            for (int st = 0; st < 2; ++st) {
                v16h kf;   // B frag: col = key (l16), k = d = st*32 + e + 16*half
                const size_t kb = ((size_t)(b * SEQ + t0 + kt * 16 + l16) * N_HEADS + h) * HEAD_DIM
                                  + st * 32 + 16 * half;
                #pragma unroll
                for (int e = 0; e < 16; ++e) kf[e] = Km[kb + e];
                a = __builtin_amdgcn_wmma_f32_16x16x32_f16(false, qf[st], false, kf,
                                                           (short)0, a, false, false);
            }
            s[kt] = a * 0.125f;
        }
        // ---- online softmax (row r lives in lanes of one half-wave) ----
        #pragma unroll
        for (int r = 0; r < 8; ++r) {
            float mx = fmaxf(s[0][r], s[1][r]);
            #pragma unroll
            for (int off = 1; off < 16; off <<= 1) mx = fmaxf(mx, __shfl_xor(mx, off, 32));
            float mnew = fmaxf(mrow[r], mx);
            float corr = __expf(mrow[r] - mnew);
            float p0 = __expf(s[0][r] - mnew);
            float p1 = __expf(s[1][r] - mnew);
            float rs = p0 + p1;
            #pragma unroll
            for (int off = 1; off < 16; off <<= 1) rs += __shfl_xor(rs, off, 32);
            lrow[r] = lrow[r] * corr + rs;
            mrow[r] = mnew;
            #pragma unroll
            for (int ni = 0; ni < 4; ++ni) o_acc[ni][r] = o_acc[ni][r] * corr;
            int prow = r + 8 * half;                      // C-layout row
            Pl[wave][prow * 32 + l16]      = (_Float16)p0;
            Pl[wave][prow * 32 + 16 + l16] = (_Float16)p1;
        }
        // ---- P (16x32) back as A-frag ----
        v16h pf;
        #pragma unroll
        for (int j = 0; j < 16; ++j) {
            int kk = j + 16 * half;
            pf[j] = Pl[wave][l16 * 32 + kk];
        }
        // ---- PV: O += P * V   (B frag contiguous from Vt) ----
        #pragma unroll
        for (int ni = 0; ni < 4; ++ni) {
            v16h vf;   // col = d (ni*16+l16), k = key = e + 16*half
            const _Float16* vp = &Vt[(ni * 16 + l16) * 32 + 16 * half];
            #pragma unroll
            for (int e = 0; e < 16; ++e) vf[e] = vp[e];
            o_acc[ni] = __builtin_amdgcn_wmma_f32_16x16x32_f16(false, pf, false, vf,
                                                               (short)0, o_acc[ni], false, false);
        }
        __syncthreads();            // Vt reused next iteration
    }
    // ---- normalize & store ----
    #pragma unroll
    for (int ni = 0; ni < 4; ++ni)
        #pragma unroll
        for (int r = 0; r < 8; ++r) {
            int row = qm0 + r + 8 * half;
            float v = o_acc[ni][r] / lrow[r];
            O[((size_t)(b * SEQ + row) * N_HEADS + h) * HEAD_DIM + ni * 16 + l16] = (_Float16)v;
        }
}

// =====================================================================
// host-side orchestration
// =====================================================================
extern "C" void kernel_launch(void* const* d_in, const int* in_sizes, int n_in,
                              void* d_out, int out_size, void* d_ws, size_t ws_size,
                              hipStream_t stream) {
    const float* x       = (const float*)d_in[0];
    const float* te      = (const float*)d_in[1];
    const float* w_norm1 = (const float*)d_in[2];
    const float* t1_w1   = (const float*)d_in[3];
    const float* t1_b1   = (const float*)d_in[4];
    const float* t1_w2   = (const float*)d_in[5];
    const float* t1_b2   = (const float*)d_in[6];
    const float* w_norm2 = (const float*)d_in[7];
    const float* t2_w1   = (const float*)d_in[8];
    const float* t2_b1   = (const float*)d_in[9];
    const float* t2_w2   = (const float*)d_in[10];
    const float* t2_b2   = (const float*)d_in[11];
    const float* wq = (const float*)d_in[12]; const float* bq = (const float*)d_in[13];
    const float* wk = (const float*)d_in[14]; const float* bk = (const float*)d_in[15];
    const float* wv = (const float*)d_in[16]; const float* bv = (const float*)d_in[17];
    const float* wo = (const float*)d_in[18]; const float* bo = (const float*)d_in[19];
    const float* w_gate = (const float*)d_in[20];
    const float* w_up   = (const float*)d_in[21];
    const float* w_down = (const float*)d_in[22];

    // ---- workspace bump allocator (256B aligned) ----
    size_t off = 0;
    auto alloc = [&](size_t bytes) -> void* {
        void* p = (char*)d_ws + off;
        off += (bytes + 255) & ~(size_t)255;
        return p;
    };
    float*     th      = (float*)    alloc(2 * BATCH * D_MODEL * 4);
    float*     ssb     = (float*)    alloc(2 * BATCH * 2 * D_MODEL * 4);
    _Float16*  wq_t    = (_Float16*) alloc((size_t)D_MODEL * D_MODEL * 2);
    _Float16*  wk_t    = (_Float16*) alloc((size_t)D_MODEL * D_MODEL * 2);
    _Float16*  wv_t    = (_Float16*) alloc((size_t)D_MODEL * D_MODEL * 2);
    _Float16*  wo_t    = (_Float16*) alloc((size_t)D_MODEL * D_MODEL * 2);
    _Float16*  wg_t    = (_Float16*) alloc((size_t)D_MODEL * D_FF * 2);
    _Float16*  wu_t    = (_Float16*) alloc((size_t)D_MODEL * D_FF * 2);
    _Float16*  wd_t    = (_Float16*) alloc((size_t)D_FF * D_MODEL * 2);
    _Float16*  nx_h    = (_Float16*) alloc((size_t)MROWS * D_MODEL * 2);
    float*     q32     = (float*)    alloc((size_t)MROWS * D_MODEL * 4);
    float*     k32     = (float*)    alloc((size_t)MROWS * D_MODEL * 4);
    _Float16*  q_h     = (_Float16*) alloc((size_t)MROWS * D_MODEL * 2);
    _Float16*  k_h     = (_Float16*) alloc((size_t)MROWS * D_MODEL * 2);
    _Float16*  v_h     = (_Float16*) alloc((size_t)MROWS * D_MODEL * 2);
    _Float16*  attn_h  = (_Float16*) alloc((size_t)MROWS * D_MODEL * 2);
    float*     x1      = (float*)    alloc((size_t)MROWS * D_MODEL * 4);
    float*     gate32  = (float*)    alloc((size_t)MROWS * D_FF * 4);
    _Float16*  ffh     = (_Float16*) alloc((size_t)MROWS * D_FF * 2);
    (void)ws_size; (void)n_in; (void)in_sizes; (void)out_size;

    // ---- 1. weights -> fp16, transposed to [N][K] for TDM-friendly B tiles ----
    auto cvt_t = [&](const float* src, _Float16* dst, int R, int C) {
        dim3 g(C / 32, R / 32);
        f32_to_f16_t_kernel<<<g, 256, 0, stream>>>(src, dst, R, C);
    };
    cvt_t(wq, wq_t, D_MODEL, D_MODEL);   cvt_t(wk, wk_t, D_MODEL, D_MODEL);
    cvt_t(wv, wv_t, D_MODEL, D_MODEL);   cvt_t(wo, wo_t, D_MODEL, D_MODEL);
    cvt_t(w_gate, wg_t, D_MODEL, D_FF);  cvt_t(w_up, wu_t, D_MODEL, D_FF);
    cvt_t(w_down, wd_t, D_FF, D_MODEL);

    // ---- 2. time MLP ----
    time_hidden_kernel<<<(2 * BATCH * D_MODEL + 255) / 256, 256, 0, stream>>>(
        te, t1_w1, t1_b1, t2_w1, t2_b1, th);
    time_ss_kernel<<<(2 * BATCH * 2 * D_MODEL + 255) / 256, 256, 0, stream>>>(
        th, t1_w2, t1_b2, t2_w2, t2_b2, ssb);

    // ---- 3. norm1 ----
    ada_norm_kernel<<<MROWS, 256, 0, stream>>>(x, w_norm1, ssb, nx_h);

    // ---- 4. QKV projections ----
    dim3 gqkv(D_MODEL / BN, MROWS / BM);
    wmma_gemm_kernel<0><<<gqkv, 256, 0, stream>>>(nx_h, wq_t, bq, nullptr, q32, nullptr,
                                                  MROWS, D_MODEL, D_MODEL);
    wmma_gemm_kernel<0><<<gqkv, 256, 0, stream>>>(nx_h, wk_t, bk, nullptr, k32, nullptr,
                                                  MROWS, D_MODEL, D_MODEL);
    wmma_gemm_kernel<2><<<gqkv, 256, 0, stream>>>(nx_h, wv_t, bv, nullptr, nullptr, v_h,
                                                  MROWS, D_MODEL, D_MODEL);

    // ---- 5. RoPE ----
    int nrope = BATCH * SEQ * N_HEADS * 32;
    rope_kernel<<<(nrope + 255) / 256, 256, 0, stream>>>(q32, k32, q_h, k_h, nrope);

    // ---- 6. attention ----
    dim3 ga(SEQ / 64, N_HEADS, BATCH);
    flash_attn_kernel<<<ga, 128, 0, stream>>>(q_h, k_h, v_h, attn_h);

    // ---- 7. O projection + residual ----
    wmma_gemm_kernel<1><<<gqkv, 256, 0, stream>>>(attn_h, wo_t, bo, x, x1, nullptr,
                                                  MROWS, D_MODEL, D_MODEL);

    // ---- 8. norm2 ----
    ada_norm_kernel<<<MROWS, 256, 0, stream>>>(x1, w_norm2, ssb + 2 * BATCH * D_MODEL, nx_h);

    // ---- 9. FFN ----
    dim3 gff(D_FF / BN, MROWS / BM);
    wmma_gemm_kernel<0><<<gff, 256, 0, stream>>>(nx_h, wg_t, nullptr, nullptr, gate32, nullptr,
                                                 MROWS, D_FF, D_MODEL);
    wmma_gemm_kernel<3><<<gff, 256, 0, stream>>>(nx_h, wu_t, nullptr, gate32, nullptr, ffh,
                                                 MROWS, D_FF, D_MODEL);
    dim3 gdn(D_MODEL / BN, MROWS / BM);
    wmma_gemm_kernel<1><<<gdn, 256, 0, stream>>>(ffh, wd_t, nullptr, x1, (float*)d_out, nullptr,
                                                 MROWS, D_MODEL, D_FF);
}
